// MultiHeadSelector_69621419868985
// MI455X (gfx1250) — compile-verified, matrix operand order checked by reading, not run on hardware
//
#include <hip/hip_runtime.h>

// ---- problem constants ----
static constexpr int kB  = 32;    // batch
static constexpr int kC  = 12;    // heads
static constexpr int kS  = 784;   // seq (patches)
static constexpr int kS1 = 785;   // seq+1
static constexpr int kH  = 768;   // hidden
static constexpr int kD  = 512;   // inner dim
static constexpr int kPN = 84;    // PATCH_NUM
static constexpr int kHG = 28;    // grid side
static constexpr int kKP = 800;   // K padded to multiple of 32 (zeros in tail)

typedef __attribute__((ext_vector_type(16))) __bf16 v16bf;
typedef __attribute__((ext_vector_type(8)))  float  v8f;

// gfx1250 async copy: global -> LDS, 16B per lane, tracked by ASYNCcnt.
// GVS form: LDS address from a VGPR; global address = SGPR base + 32-bit VGPR offset
__device__ __forceinline__ void async_b128(uint32_t lds_off, const void* sbase,
                                           uint32_t voff) {
    asm volatile("global_load_async_to_lds_b128 %0, %1, %2"
                 :: "v"(lds_off), "v"(voff), "s"(sbase)
                 : "memory");
}

// ---------------------------------------------------------------------------
// pos_w[b][n][m] = mean_c x[b][c][1+n][1+m]  -> bf16 (WMMA A-operand)
// ---------------------------------------------------------------------------
__global__ void k_posw(const float* __restrict__ x, __bf16* __restrict__ pw) {
    int i = blockIdx.x * 256 + threadIdx.x;
    const int total = kB * kS * kS;
    if (i >= total) return;
    int m  = i % kS;
    int bn = i / kS;
    int n  = bn % kS;
    int b  = bn / kS;
    const float* p = x + (((size_t)b * kC) * kS1 + (n + 1)) * kS1 + (m + 1);
    float s = 0.f;
#pragma unroll
    for (int c = 0; c < kC; ++c) s += p[(size_t)c * kS1 * kS1];
    pw[i] = (__bf16)(s * (1.f / 12.f));
}

// ---------------------------------------------------------------------------
// t1t[b][h][m] = sum_c score[b][c][m] * w1[c][h], K-major, K padded to 800
// (tail columns written as exact zeros so the GEMM K-tail is inert)
// ---------------------------------------------------------------------------
__global__ void k_t1t(const float* __restrict__ x, const float* __restrict__ w1,
                      __bf16* __restrict__ t1t) {
    int i = blockIdx.x * 256 + threadIdx.x;
    const int total = kB * kD * kKP;
    if (i >= total) return;
    int m = i % kKP;
    int h = (i / kKP) % kD;
    int b = i / (kKP * kD);
    float s = 0.f;
    if (m < kS) {
#pragma unroll
        for (int c = 0; c < kC; ++c)
            s += x[(((size_t)b * kC + c) * kS1) * kS1 + (m + 1)] * w1[c * kD + h];
    }
    t1t[i] = (__bf16)s;
}

__global__ void k_zero(float* __restrict__ p, int n) {
    int i = blockIdx.x * 256 + threadIdx.x;
    if (i < n) p[i] = 0.f;
}

// ---------------------------------------------------------------------------
// counts[b][i] += 1 for each head where i is in top-84 of score[b][c][:]
// ---------------------------------------------------------------------------
__global__ void k_counts(const float* __restrict__ x, float* __restrict__ counts) {
    __shared__ float sc[kS];
    int bc = blockIdx.x;
    int b = bc / kC, c = bc % kC;
    const float* row = x + (((size_t)b * kC + c) * kS1) * kS1 + 1;
    for (int i = threadIdx.x; i < kS; i += blockDim.x) sc[i] = row[i];
    __syncthreads();
    for (int i = threadIdx.x; i < kS; i += blockDim.x) {
        float vi = sc[i];
        int r = 0;
        for (int j = 0; j < kS; ++j) {
            float vj = sc[j];
            r += (vj > vi) || (vj == vi && j < i);
        }
        if (r < kPN) atomicAdd(&counts[b * kS + i], 1.0f);
    }
}

// ---------------------------------------------------------------------------
// basic_index[b] = argmax_n sum_c score[b][c][n]  (first occurrence on ties)
// ---------------------------------------------------------------------------
__global__ void k_bidx(const float* __restrict__ x, int* __restrict__ bidx) {
    __shared__ float sv[kS];
    int b = blockIdx.x;
    for (int n = threadIdx.x; n < kS; n += blockDim.x) {
        float s = 0.f;
#pragma unroll
        for (int c = 0; c < kC; ++c)
            s += x[(((size_t)b * kC + c) * kS1) * kS1 + 1 + n];
        sv[n] = s;
    }
    __syncthreads();
    if (threadIdx.x == 0) {
        int bi = 0;
        float best = sv[0];
        for (int n = 1; n < kS; ++n)
            if (sv[n] > best) { best = sv[n]; bi = n; }
        bidx[b] = bi;
    }
}

// ---------------------------------------------------------------------------
// 3x3 weighted conv of counts, then stable descending rank -> patch_idx
// ---------------------------------------------------------------------------
__global__ void k_order(const float* __restrict__ counts,
                        float* __restrict__ pidx_f, int* __restrict__ pidx_i) {
    __shared__ float sc[kS];
    int b = blockIdx.x;
    for (int n = threadIdx.x; n < kS; n += blockDim.x) {
        int y = n / kHG, xx = n % kHG;
        float s = 0.f;
#pragma unroll
        for (int dy = -1; dy <= 1; ++dy)
#pragma unroll
            for (int dx = -1; dx <= 1; ++dx) {
                int yy = y + dy, xc = xx + dx;
                if (yy >= 0 && yy < kHG && xc >= 0 && xc < kHG) {
                    float w = ((dy == 0) ? 2.f : 1.f) * ((dx == 0) ? 2.f : 1.f);
                    s += w * counts[b * kS + yy * kHG + xc];
                }
            }
        sc[n] = s;
    }
    __syncthreads();
    for (int i = threadIdx.x; i < kS; i += blockDim.x) {
        float vi = sc[i];
        int r = 0;
        for (int j = 0; j < kS; ++j) {
            float vj = sc[j];
            r += (vj > vi) || (vj == vi && j < i);
        }
        if (r < kPN) {
            pidx_f[b * kPN + r] = (float)(i + 1);
            pidx_i[b * kPN + r] = i + 1;
        }
    }
}

// ---------------------------------------------------------------------------
// g1[b] = relu(pos_w[b] @ t1[b]) : M=784, N=512, K=784(+pad), bf16 WMMA
// 256 threads = 8 waves (2x4); wave tile 32x32; block tile 64x128.
// Double-buffered LDS fed by GLOBAL_LOAD_ASYNC_TO_LDS_B128 (ASYNCcnt).
// No load guards: B K-tail is zero-padded; A tail reads stay inside d_ws and
// are either multiplied by zero or land in unstored output rows.
// ---------------------------------------------------------------------------
__global__ void __launch_bounds__(256)
k_gemm1(const __bf16* __restrict__ pw, const __bf16* __restrict__ t1t,
        float* __restrict__ g1) {
    // rows padded to 40 bf16 (80B): 16B-aligned b128 frags, bank spreading
    __shared__ __align__(16) __bf16 lA[2][64 * 40];
    __shared__ __align__(16) __bf16 lB[2][128 * 40];

    const int b    = blockIdx.z;
    const int m0   = blockIdx.y * 64;
    const int n0   = blockIdx.x * 128;
    const int t    = threadIdx.x;
    const int lane = t & 31;
    const int wave = t >> 5;
    const int wm   = (wave & 1) * 32;
    const int wn   = (wave >> 1) * 32;
    const int frRow = lane & 15;
    const int kb    = (lane >> 4) * 8;   // K sub-offset per ISA 16-bit layout

    const __bf16* Ab = pw  + (size_t)b * kS * kS;   // row stride 784 bf16
    const __bf16* Bb = t1t + (size_t)b * kD * kKP;  // row stride 800 bf16

    // staging coords: A 64x32 -> 1 b128/thread; B 128x32 -> 2 b128/thread
    const int arow = t >> 2, ach = (t & 3) * 8;
    const uint32_t avoff = (uint32_t)((m0 + arow) * (kS * 2) + ach * 2);
    const int brow0 = t >> 2,          bch0 = (t & 3) * 8;
    const int brow1 = (t + 256) >> 2,  bch1 = ((t + 256) & 3) * 8;
    const uint32_t bvoff0 = (uint32_t)((n0 + brow0) * (kKP * 2) + bch0 * 2);
    const uint32_t bvoff1 = (uint32_t)((n0 + brow1) * (kKP * 2) + bch1 * 2);

    v8f acc[2][2];
#pragma unroll
    for (int i = 0; i < 2; ++i)
#pragma unroll
        for (int j = 0; j < 2; ++j)
#pragma unroll
            for (int e = 0; e < 8; ++e) acc[i][j][e] = 0.f;

    union Frag { v16bf v; uint4 q[2]; };

    auto issue = [&](int bufi, int k0) {
        async_b128((uint32_t)(uintptr_t)&lA[bufi][arow * 40 + ach],  Ab,
                   avoff + (uint32_t)(k0 * 2));
        async_b128((uint32_t)(uintptr_t)&lB[bufi][brow0 * 40 + bch0], Bb,
                   bvoff0 + (uint32_t)(k0 * 2));
        async_b128((uint32_t)(uintptr_t)&lB[bufi][brow1 * 40 + bch1], Bb,
                   bvoff1 + (uint32_t)(k0 * 2));
    };

    const int kSteps = kKP / 32;  // 25
    int buf = 0;
    issue(0, 0);

    for (int it = 0; it < kSteps; ++it) {
        if (it + 1 < kSteps) {
            issue(buf ^ 1, (it + 1) * 32);
            // 3 next-buffer loads outstanding; current buffer resident
            asm volatile("s_wait_asynccnt 3" ::: "memory");
        } else {
            asm volatile("s_wait_asynccnt 0" ::: "memory");
        }
        __syncthreads();

        Frag a[2], bf[2];
#pragma unroll
        for (int mt = 0; mt < 2; ++mt) {
            int r = wm + mt * 16 + frRow;
            a[mt].q[0] = *(const uint4*)&lA[buf][r * 40 + kb];
            a[mt].q[1] = *(const uint4*)&lA[buf][r * 40 + 16 + kb];
        }
#pragma unroll
        for (int nt = 0; nt < 2; ++nt) {
            int r = wn + nt * 16 + frRow;
            bf[nt].q[0] = *(const uint4*)&lB[buf][r * 40 + kb];
            bf[nt].q[1] = *(const uint4*)&lB[buf][r * 40 + 16 + kb];
        }
#pragma unroll
        for (int mt = 0; mt < 2; ++mt)
#pragma unroll
            for (int nt = 0; nt < 2; ++nt)
                acc[mt][nt] = __builtin_amdgcn_wmma_f32_16x16x32_bf16(
                    false, a[mt].v, false, bf[nt].v,
                    (short)0, acc[mt][nt], false, false);

        __syncthreads();  // all reads done before next issue overwrites buf
        buf ^= 1;
    }

    // store with relu; D layout: lane<16 -> M=r, lane>=16 -> M=8+r; N=lane&15
#pragma unroll
    for (int mt = 0; mt < 2; ++mt) {
#pragma unroll
        for (int nt = 0; nt < 2; ++nt) {
            int n = n0 + wn + nt * 16 + (lane & 15);
            int mbase = m0 + wm + mt * 16 + ((lane >> 4) * 8);
#pragma unroll
            for (int r = 0; r < 8; ++r) {
                int m = mbase + r;
                if (m < kS)
                    g1[((size_t)b * kS + m) * kD + n] = fmaxf(acc[mt][nt][r], 0.f);
            }
        }
    }
}

// ---------------------------------------------------------------------------
// v[b][j] = sum_m pos_row[m] * g1[b][m][j]   (pos_row recomputed f32)
// ---------------------------------------------------------------------------
__global__ void k_gemv(const float* __restrict__ x, const int* __restrict__ bidx,
                       const float* __restrict__ g1, float* __restrict__ v) {
    __shared__ float pr[kS];
    int b = blockIdx.x;
    int bi = bidx[b];
    for (int m = threadIdx.x; m < kS; m += blockDim.x) {
        float s = 0.f;
#pragma unroll
        for (int c = 0; c < kC; ++c)
            s += x[(((size_t)b * kC + c) * kS1 + 1 + bi) * kS1 + 1 + m];
        pr[m] = s * (1.f / 12.f);
    }
    __syncthreads();
    int j = threadIdx.x;  // blockDim == 512
    float acc = 0.f;
    const float* gb = g1 + (size_t)b * kS * kD + j;
    for (int m = 0; m < kS; ++m) acc += pr[m] * gb[(size_t)m * kD];
    v[b * kD + j] = acc;
}

// ---------------------------------------------------------------------------
// add[b][h] = lrelu( sum_k v[b][k] * w2[k][h] )
// ---------------------------------------------------------------------------
__global__ void k_add(const float* __restrict__ v, const float* __restrict__ w2,
                      float* __restrict__ add) {
    __shared__ float sv[kD];
    int b = blockIdx.x;
    int h = threadIdx.x;  // blockDim == 768
    for (int kx = threadIdx.x; kx < kD; kx += blockDim.x) sv[kx] = v[b * kD + kx];
    __syncthreads();
    float a = 0.f;
    for (int kx = 0; kx < kD; ++kx) a += sv[kx] * w2[kx * kH + h];
    add[b * kH + h] = (a > 0.f) ? a : 0.2f * a;
}

// ---------------------------------------------------------------------------
// hs_out = hidden_states, row 0 += add
// ---------------------------------------------------------------------------
__global__ void k_hs(const float* __restrict__ hs, const float* __restrict__ add,
                     float* __restrict__ out) {
    int i = blockIdx.x * 256 + threadIdx.x;
    const int total = kB * kS1 * kH;
    if (i >= total) return;
    int h = i % kH;
    int s = (i / kH) % kS1;
    int b = i / (kH * kS1);
    float val = hs[i];
    if (s == 0) val += add[b * kH + h];
    out[i] = val;
}

// ---------------------------------------------------------------------------
// selected[b][p][:] = hs_out[b][patch_idx[b][p]][:]  (idx >= 1, no add term)
// ---------------------------------------------------------------------------
__global__ void k_sel(const float* __restrict__ hs, const int* __restrict__ pidx,
                      float* __restrict__ out) {
    int i = blockIdx.x * 256 + threadIdx.x;
    const int total = kB * kPN * kH;
    if (i >= total) return;
    int h = i % kH;
    int p = (i / kH) % kPN;
    int b = i / (kH * kPN);
    int idx = pidx[b * kPN + p];
    out[i] = hs[((size_t)b * kS1 + idx) * kH + h];
}

// ---------------------------------------------------------------------------
extern "C" void kernel_launch(void* const* d_in, const int* in_sizes, int n_in,
                              void* d_out, int out_size, void* d_ws, size_t ws_size,
                              hipStream_t stream) {
    (void)in_sizes; (void)n_in; (void)out_size; (void)ws_size;
    const float* hs = (const float*)d_in[0];
    const float* x  = (const float*)d_in[1];
    // d_in[2] = contribution (unused by the reference)
    const float* w1 = (const float*)d_in[3];
    const float* w2 = (const float*)d_in[4];
    float* out = (float*)d_out;

    // workspace carve-up (~118 MB)
    char* ws = (char*)d_ws;
    size_t off = 0;
    auto alloc = [&](size_t bytes) -> char* {
        char* p = ws + off;
        off += (bytes + 255) & ~(size_t)255;
        return p;
    };
    __bf16* pw     = (__bf16*)alloc((size_t)kB * kS * kS * 2);
    __bf16* t1t    = (__bf16*)alloc((size_t)kB * kD * kKP * 2);
    float*  g1     = (float*)alloc((size_t)kB * kS * kD * 4);
    float*  counts = (float*)alloc((size_t)kB * kS * 4);
    int*    bidx   = (int*)alloc((size_t)kB * 4);
    float*  v      = (float*)alloc((size_t)kB * kD * 4);
    float*  addv   = (float*)alloc((size_t)kB * kH * 4);
    int*    pidx   = (int*)alloc((size_t)kB * kPN * 4);

    float* out_hs   = out;
    float* out_sel  = out + (size_t)kB * kS1 * kH;
    float* out_pidx = out_sel + (size_t)kB * kPN * kH;

    k_posw<<<(kB * kS * kS + 255) / 256, 256, 0, stream>>>(x, pw);
    k_t1t<<<(kB * kD * kKP + 255) / 256, 256, 0, stream>>>(x, w1, t1t);
    k_zero<<<(kB * kS + 255) / 256, 256, 0, stream>>>(counts, kB * kS);
    k_counts<<<kB * kC, 256, 0, stream>>>(x, counts);
    k_bidx<<<kB, 256, 0, stream>>>(x, bidx);
    k_order<<<kB, 256, 0, stream>>>(counts, out_pidx, pidx);
    k_gemm1<<<dim3(kD / 128, (kS + 63) / 64, kB), 256, 0, stream>>>(pw, t1t, g1);
    k_gemv<<<kB, 512, 0, stream>>>(x, bidx, g1, v);
    k_add<<<kB, kH, 0, stream>>>(v, w2, addv);
    k_hs<<<(kB * kS1 * kH + 255) / 256, 256, 0, stream>>>(hs, addv, out_hs);
    k_sel<<<(kB * kPN * kH + 255) / 256, 256, 0, stream>>>(hs, pidx, out_sel);
}
